// tree_attention_abstract_DP_1460288880927
// MI455X (gfx1250) — matrix operand order repaired and checked by reading, not compiled, not run on hardware
//
#include <hip/hip_runtime.h>
#include <hip/hip_bf16.h>
#include <math.h>

// ---------------------------------------------------------------------------
// Types / helpers for CDNA5 WMMA (wave32, v_wmma_f32_16x16x32_bf16)
// ---------------------------------------------------------------------------
typedef __attribute__((ext_vector_type(16))) __bf16 v16bf;
typedef __attribute__((ext_vector_type(8)))  __bf16 v8bf;
typedef __attribute__((ext_vector_type(8)))  float  v8f;

#define DEV __device__ __forceinline__

DEV v16bf make_frag(const __bf16* p0, const __bf16* p1) {
  v8bf lo = *(const v8bf*)p0;
  v8bf hi = *(const v8bf*)p1;
  return __builtin_shufflevector(lo, hi, 0,1,2,3,4,5,6,7,8,9,10,11,12,13,14,15);
}

DEV v8f wmma_bf16(v16bf a, v16bf b, v8f c) {
  // D = A x B + C, 16x16x32 bf16 -> f32
  return __builtin_amdgcn_wmma_f32_16x16x32_bf16(false, a, false, b, (short)0, c,
                                                 false, false);
}

DEV float sigm(float x) { return 1.0f / (1.0f + expf(-x)); }

DEV float block_sum(float v, float* sm) {
  int tid = threadIdx.x;
  sm[tid] = v; __syncthreads();
  for (int s = blockDim.x >> 1; s > 0; s >>= 1) {
    if (tid < s) sm[tid] += sm[tid + s];
    __syncthreads();
  }
  float r = sm[0]; __syncthreads();
  return r;
}

// ---------------------------------------------------------------------------
// Problem constants
// ---------------------------------------------------------------------------
#define BATCH   128
#define TSEQ    45
#define EMB     300
#define EMB_P   320      // K padded to multiple of 32
#define HID     1024
#define G4      4096     // 4*HID
#define HW      14
#define PIX     (BATCH*HW*HW)   // 25088 output pixels
#define C0IN    1026
#define C0IN_P  1056     // padded to multiple of 32
#define CCAT    130
#define CCAT_P  160
#define CH      128

// ---------------------------------------------------------------------------
// Utility kernels
// ---------------------------------------------------------------------------
__global__ void k_zero(uint4* p, long n16) {
  long i = (long)blockIdx.x * blockDim.x + threadIdx.x;
  long stride = (long)gridDim.x * blockDim.x;
  uint4 z = {0u, 0u, 0u, 0u};
  for (; i < n16; i += stride) p[i] = z;
}

// emb = lookup[que] transposed to [T,B,E], stored bf16 padded [T*B][EMB_P]
__global__ void k_embed(const int* que, const float* lookup, __bf16* X) {
  long idx = (long)blockIdx.x * blockDim.x + threadIdx.x;
  if (idx >= (long)TSEQ * BATCH * EMB) return;
  int k   = idx % EMB;
  int row = idx / EMB;          // t*B + b
  int t = row / BATCH, b = row % BATCH;
  int tok = que[b * TSEQ + t];
  X[(long)row * EMB_P + k] = (__bf16)lookup[(long)tok * EMB + k];
}

// dense f32 [N][K] -> bf16 [N][ldd] (pad columns untouched: pre-zeroed)
__global__ void k_cvt_pad(const float* src, __bf16* dst, int N, int K, int ldd) {
  long idx = (long)blockIdx.x * blockDim.x + threadIdx.x;
  if (idx >= (long)N * K) return;
  int n = idx / K, k = idx % K;
  dst[(long)n * ldd + k] = (__bf16)src[idx];
}

// conv weight [128][Cin][3][3] -> bf16 [tap][128][ldc]
__global__ void k_cvt_convw(const float* w, __bf16* dst, int Cin, int ldc) {
  long idx = (long)blockIdx.x * blockDim.x + threadIdx.x;
  long total = (long)CH * Cin * 9;
  if (idx >= total) return;
  int n   = idx / (Cin * 9);
  int rem = idx % (Cin * 9);
  int c   = rem / 9;
  int tap = rem % 9;
  dst[((long)tap * CH + n) * ldc + c] = (__bf16)w[idx];
}

__global__ void k_addb(const float* a, const float* b, float* o, int n) {
  int i = blockIdx.x * blockDim.x + threadIdx.x;
  if (i < n) o[i] = a[i] + b[i];
}

// ---------------------------------------------------------------------------
// Generic bf16 GEMM: C[M][ldc] = A[M][lda] * W[N][ldb]^T (+bias, relu)
// block = 128 threads (4 waves); wave -> 16 rows x 64 cols
// grid = (M/64, N/64)
// ---------------------------------------------------------------------------
__global__ __launch_bounds__(128)
void k_gemm(const __bf16* A, int lda, const __bf16* W, int ldb,
            float* C, int ldc, const float* bias, int relu, int K) {
  int lane = threadIdx.x & 31;
  int wave = threadIdx.x >> 5;
  int mtile = blockIdx.x * 4 + wave;
  int nbase = blockIdx.y * 64;
  const __bf16* Arow = A + (long)(mtile * 16 + (lane & 15)) * lda;
  int aoff = (lane & 16) ? 16 : 0;
  int boff = (lane & 16) ? 16 : 0;
  const __bf16* Wrow[4];
#pragma unroll
  for (int j = 0; j < 4; j++)
    Wrow[j] = W + (long)(nbase + j * 16 + (lane & 15)) * ldb + boff;
  v8f acc[4] = {};
  for (int kb = 0; kb < K; kb += 32) {
    v16bf a = make_frag(Arow + kb + aoff, Arow + kb + 8 + aoff);
#pragma unroll
    for (int j = 0; j < 4; j++) {
      v16bf b = *(const v16bf*)(Wrow[j] + kb);
      acc[j] = wmma_bf16(a, b, acc[j]);
    }
  }
  int mrow = mtile * 16 + ((lane & 16) ? 8 : 0);
  int n0   = lane & 15;
#pragma unroll
  for (int j = 0; j < 4; j++) {
    int n = nbase + j * 16 + n0;
    float bz = bias ? bias[n] : 0.0f;
#pragma unroll
    for (int r = 0; r < 8; r++) {
      float v = acc[j][r] + bz;
      if (relu) v = fmaxf(v, 0.0f);
      C[(long)(mrow + r) * ldc + n] = v;
    }
  }
}

// ---------------------------------------------------------------------------
// Fused LSTM step: gates = hprev@Whh^T + Xg[t] (bias pre-folded into Xg),
// then c,h update. Wave computes the same 16 hidden cols for all 4 gates.
// grid = (64, 2), block = 128.
// ---------------------------------------------------------------------------
__global__ __launch_bounds__(128)
void k_lstm_step(const __bf16* hprev, const __bf16* Whh, const float* Xg_t,
                 float* cbuf, __bf16* hnext, float* hf32) {
  int lane = threadIdx.x & 31;
  int wave = threadIdx.x >> 5;
  int ct = blockIdx.x;                 // hidden col tile (16 cols)
  int mt = blockIdx.y * 4 + wave;      // row tile (16 rows)
  const __bf16* Arow = hprev + (long)(mt * 16 + (lane & 15)) * HID;
  int aoff = (lane & 16) ? 16 : 0;
  int boff = (lane & 16) ? 16 : 0;
  const __bf16* Wrow[4];
#pragma unroll
  for (int g = 0; g < 4; g++)
    Wrow[g] = Whh + (long)(g * HID + ct * 16 + (lane & 15)) * HID + boff;
  v8f acc[4] = {};
  for (int kb = 0; kb < HID; kb += 32) {
    v16bf a = make_frag(Arow + kb + aoff, Arow + kb + 8 + aoff);
#pragma unroll
    for (int g = 0; g < 4; g++) {
      v16bf b = *(const v16bf*)(Wrow[g] + kb);
      acc[g] = wmma_bf16(a, b, acc[g]);
    }
  }
  int mrow = mt * 16 + ((lane & 16) ? 8 : 0);
  int n = ct * 16 + (lane & 15);
#pragma unroll
  for (int r = 0; r < 8; r++) {
    int m = mrow + r;
    const float* xg = Xg_t + (long)m * G4;
    float gi = acc[0][r] + xg[n];
    float gf = acc[1][r] + xg[n + HID];
    float gg = acc[2][r] + xg[n + 2 * HID];
    float go = acc[3][r] + xg[n + 3 * HID];
    float c_old = cbuf[m * HID + n];
    float cn = sigm(gf) * c_old + sigm(gi) * tanhf(gg);
    float hn = sigm(go) * tanhf(cn);
    cbuf[m * HID + n] = cn;
    hf32[m * HID + n] = hn;
    hnext[m * HID + n] = (__bf16)hn;
  }
}

// backward direction, step 0 only (h=c=0): pure elementwise from Xg_b
__global__ void k_lstm_b0(const float* Xg_b, float* hb0) {
  int idx = blockIdx.x * blockDim.x + threadIdx.x;
  if (idx >= BATCH * HID) return;
  int m = idx >> 10, n = idx & 1023;
  const float* xg = Xg_b + (long)m * G4;
  float gi = xg[n], gg = xg[n + 2 * HID], go = xg[n + 3 * HID];
  float cn = sigm(gi) * tanhf(gg);
  hb0[idx] = sigm(go) * tanhf(cn);
}

// enc = l2norm(concat(hf, hb)) per row -> d_out[0 : 128*2048]
__global__ void k_enc_norm(const float* hf, const float* hb, float* out) {
  __shared__ float sm[256];
  int b = blockIdx.x;
  float s = 0.0f;
  for (int j = threadIdx.x; j < HID; j += 256) {
    float v = hf[b * HID + j]; s += v * v;
    v = hb[b * HID + j];       s += v * v;
  }
  float tot = block_sum(s, sm);
  float inv = 1.0f / fmaxf(sqrtf(tot), 1e-12f);
  for (int j = threadIdx.x; j < HID; j += 256) {
    out[b * 2048 + j]        = hf[b * HID + j] * inv;
    out[b * 2048 + HID + j]  = hb[b * HID + j] * inv;
  }
}

// ---------------------------------------------------------------------------
// Image branch
// ---------------------------------------------------------------------------
// per-pixel channel L2 norm; writes channels-last bf16 into spatially padded
// buffer vpad[B][16][16][C0IN_P] (halo/pad pre-zeroed); appends coord channels.
__global__ void k_imgnorm(const float* img, __bf16* vpad) {
  __shared__ float sm[256];
  int p = blockIdx.x;                       // 0..PIX-1
  int b = p / 196, rem = p % 196, y = rem / 14, x = rem % 14;
  const float* base = img + (long)b * 1024 * 196 + rem;  // NCHW, ch stride 196
  float s = 0.0f;
  for (int c = threadIdx.x; c < 1024; c += 256) {
    float v = base[(long)c * 196]; s += v * v;
  }
  float tot = block_sum(s, sm);
  float inv = 1.0f / fmaxf(sqrtf(tot), 1e-12f);
  __bf16* o = vpad + (((long)b * 16 + (y + 1)) * 16 + (x + 1)) * C0IN_P;
  for (int c = threadIdx.x; c < 1024; c += 256)
    o[c] = (__bf16)(base[(long)c * 196] * inv);
  if (threadIdx.x == 0) {
    o[1024] = (__bf16)((y - 7) / 7.0f);
    o[1025] = (__bf16)((x - 7) / 7.0f);
  }
}

// 3x3 conv as implicit GEMM over 9 taps. inpad[B][16][16][CHN] bf16,
// wt[9][128][CHN] bf16, out[PIX][128] f32. block=128 (4 waves), wave=16 pix.
__global__ __launch_bounds__(128)
void k_conv3x3(const __bf16* inpad, int CHN, const __bf16* wt,
               const float* bias, float* out) {
  int lane = threadIdx.x & 31;
  int wave = threadIdx.x >> 5;
  int mtile = blockIdx.x * 4 + wave;
  int p = mtile * 16 + (lane & 15);
  int b = p / 196, rem = p % 196, y = rem / 14, x = rem % 14;
  const __bf16* Abase = inpad + (((long)b * 16 + y) * 16 + x) * CHN;
  int aoff = (lane & 16) ? 16 : 0;
  int boff = (lane & 16) ? 16 : 0;
  v8f acc[8] = {};
  for (int tap = 0; tap < 9; tap++) {
    int ky = tap / 3, kx = tap % 3;
    const __bf16* Arow = Abase + ((long)(ky * 16 + kx)) * CHN;
    const __bf16* Wt = wt + (long)tap * CH * CHN + boff;
    for (int kb = 0; kb < CHN; kb += 32) {
      v16bf a = make_frag(Arow + kb + aoff, Arow + kb + 8 + aoff);
#pragma unroll
      for (int j = 0; j < 8; j++) {
        v16bf bf = *(const v16bf*)(Wt + (long)(j * 16 + (lane & 15)) * CHN + kb);
        acc[j] = wmma_bf16(a, bf, acc[j]);
      }
    }
  }
  int mrow = mtile * 16 + ((lane & 16) ? 8 : 0);
  int n0 = lane & 15;
#pragma unroll
  for (int j = 0; j < 8; j++) {
    int n = j * 16 + n0;
    float bz = bias ? bias[n] : 0.0f;
#pragma unroll
    for (int r = 0; r < 8; r++)
      out[(long)(mrow + r) * CH + n] = acc[j][r] + bz;
  }
}

// per-channel batch-norm stats over PIX samples (train mode, ddof=0)
__global__ void k_bnstats(const float* z, float* mean, float* var) {
  __shared__ float sm[256];
  int c = blockIdx.x;
  float s1 = 0.0f, s2 = 0.0f;
  for (int p = threadIdx.x; p < PIX; p += 256) {
    float v = z[(long)p * CH + c];
    s1 += v; s2 += v * v;
  }
  float t1 = block_sum(s1, sm);
  float t2 = block_sum(s2, sm);
  if (threadIdx.x == 0) {
    float m = t1 / (float)PIX;
    mean[c] = m;
    var[c]  = t2 / (float)PIX - m * m;
  }
}

// v = relu(bn(z)) [+ resid]; optionally emit bf16 concat [p][CCAT_P] with
// coords, and/or NCHW f32 output.
__global__ void k_bnapply(const float* z, const float* mean, const float* var,
                          const float* g, const float* bta, const float* resid,
                          __bf16* vcat, float* outNCHW) {
  long idx = (long)blockIdx.x * blockDim.x + threadIdx.x;
  if (idx >= (long)PIX * CH) return;
  int p = idx >> 7, c = idx & 127;
  float inv = rsqrtf(var[c] + 1e-5f);
  float v = (z[idx] - mean[c]) * inv * g[c] + bta[c];
  v = fmaxf(v, 0.0f);
  if (resid) v += resid[idx];
  if (vcat) {
    vcat[(long)p * CCAT_P + c] = (__bf16)v;
    if (c == 0) {
      int rem = p % 196, y = rem / 14, x = rem % 14;
      vcat[(long)p * CCAT_P + 128] = (__bf16)((y - 7) / 7.0f);
      vcat[(long)p * CCAT_P + 129] = (__bf16)((x - 7) / 7.0f);
    }
  }
  if (outNCHW) {
    int b = p / 196, rem = p % 196;
    outNCHW[((long)b * CH + c) * 196 + rem] = v;
  }
}

// pack relu'd conv1 output (already f32) into spatially padded bf16 buffer
__global__ void k_v1pad(const float* v1, __bf16* v1pad) {
  long idx = (long)blockIdx.x * blockDim.x + threadIdx.x;
  if (idx >= (long)PIX * CH) return;
  int p = idx >> 7, c = idx & 127;
  int b = p / 196, rem = p % 196, y = rem / 14, x = rem % 14;
  v1pad[(((long)b * 16 + (y + 1)) * 16 + (x + 1)) * CH + c] = (__bf16)v1[idx];
}

// ---------------------------------------------------------------------------
// Host orchestration
// ---------------------------------------------------------------------------
extern "C" void kernel_launch(void* const* d_in, const int* in_sizes, int n_in,
                              void* d_out, int out_size, void* d_ws, size_t ws_size,
                              hipStream_t stream) {
  const int*   que     = (const int*)  d_in[0];
  const float* img     = (const float*)d_in[1];
  const float* lookup  = (const float*)d_in[2];
  const float* Wih_f   = (const float*)d_in[3];
  const float* Whh_f   = (const float*)d_in[4];
  const float* bih_f   = (const float*)d_in[5];
  const float* bhh_f   = (const float*)d_in[6];
  const float* Wih_b   = (const float*)d_in[7];
  // d_in[8] = Whh_b : not needed (backward contributes only its first step, h=0)
  const float* bih_b   = (const float*)d_in[9];
  const float* bhh_b   = (const float*)d_in[10];
  const float* conv0_w = (const float*)d_in[11];
  const float* conv0_b = (const float*)d_in[12];
  const float* bn0_g   = (const float*)d_in[13];
  const float* bn0_b   = (const float*)d_in[14];
  const float* r1c1_w  = (const float*)d_in[15];
  const float* r1c1_b  = (const float*)d_in[16];
  const float* r1c2_w  = (const float*)d_in[17];
  const float* r1c2_b  = (const float*)d_in[18];
  const float* r1bn_g  = (const float*)d_in[19];
  const float* r1bn_b  = (const float*)d_in[20];
  const float* r2c1_w  = (const float*)d_in[21];
  const float* r2c1_b  = (const float*)d_in[22];
  const float* r2c2_w  = (const float*)d_in[23];
  const float* r2c2_b  = (const float*)d_in[24];
  const float* r2bn_g  = (const float*)d_in[25];
  const float* r2bn_b  = (const float*)d_in[26];

  char* ws = (char*)d_ws;
  size_t off = 0;
  auto take = [&](size_t bytes) { size_t o = off; off += bytes; return o; };

  // ---- region that must start zeroed (pads, halos, initial h/c) ----
  const size_t oX16  = take((size_t)TSEQ * BATCH * EMB_P * 2);
  const size_t oWF   = take((size_t)G4 * EMB_P * 2);
  const size_t oWB   = take((size_t)G4 * EMB_P * 2);
  const size_t oWHH  = take((size_t)G4 * HID * 2);
  const size_t oW0C  = take((size_t)9 * CH * C0IN_P * 2);
  const size_t oW1C  = take((size_t)CH * CCAT_P * 2);
  const size_t oW2C  = take((size_t)9 * CH * CH * 2);
  const size_t oVPAD = take((size_t)BATCH * 16 * 16 * C0IN_P * 2);
  const size_t oVCAT = take((size_t)PIX * CCAT_P * 2);
  const size_t oV1P  = take((size_t)BATCH * 16 * 16 * CH * 2);
  const size_t oH16A = take((size_t)BATCH * HID * 2);
  const size_t oH16B = take((size_t)BATCH * HID * 2);
  const size_t oCBUF = take((size_t)BATCH * HID * 4);
  const size_t zeroBytes = off;
  // ---- fully-overwritten buffers ----
  const size_t oBSF  = take((size_t)G4 * 4);
  const size_t oBSB  = take((size_t)G4 * 4);
  const size_t oXGF  = take((size_t)TSEQ * BATCH * G4 * 4);
  const size_t oXGB  = take((size_t)BATCH * G4 * 4);
  const size_t oHF32 = take((size_t)BATCH * HID * 4);
  const size_t oHB0  = take((size_t)BATCH * HID * 4);
  const size_t oOUTA = take((size_t)PIX * CH * 4);
  const size_t oOUTB = take((size_t)PIX * CH * 4);
  const size_t oMEAN = take((size_t)CH * 4);
  const size_t oVAR  = take((size_t)CH * 4);
  (void)ws_size; (void)in_sizes; (void)n_in; (void)out_size;

  __bf16* X16  = (__bf16*)(ws + oX16);
  __bf16* WF   = (__bf16*)(ws + oWF);
  __bf16* WB   = (__bf16*)(ws + oWB);
  __bf16* WHH  = (__bf16*)(ws + oWHH);
  __bf16* W0C  = (__bf16*)(ws + oW0C);
  __bf16* W1C  = (__bf16*)(ws + oW1C);
  __bf16* W2C  = (__bf16*)(ws + oW2C);
  __bf16* VPAD = (__bf16*)(ws + oVPAD);
  __bf16* VCAT = (__bf16*)(ws + oVCAT);
  __bf16* V1P  = (__bf16*)(ws + oV1P);
  __bf16* H16A = (__bf16*)(ws + oH16A);
  __bf16* H16B = (__bf16*)(ws + oH16B);
  float* CBUF = (float*)(ws + oCBUF);
  float* BSF  = (float*)(ws + oBSF);
  float* BSB  = (float*)(ws + oBSB);
  float* XGF  = (float*)(ws + oXGF);
  float* XGB  = (float*)(ws + oXGB);
  float* HF32 = (float*)(ws + oHF32);
  float* HB0  = (float*)(ws + oHB0);
  float* OUTA = (float*)(ws + oOUTA);
  float* OUTB = (float*)(ws + oOUTB);
  float* MEAN = (float*)(ws + oMEAN);
  float* VAR  = (float*)(ws + oVAR);

  float* enc_out = (float*)d_out;                 // [128,2048]
  float* v_out   = (float*)d_out + 128 * 2048;    // [128,128,14,14] NCHW

  // 0) zero pads / halos / initial recurrent state
  k_zero<<<4096, 256, 0, stream>>>((uint4*)ws, (long)(zeroBytes / 16));

  // 1) conversions + embedding
  {
    long n = (long)TSEQ * BATCH * EMB;
    k_embed<<<(int)((n + 255) / 256), 256, 0, stream>>>(que, lookup, X16);
  }
  { long n = (long)G4 * EMB;  k_cvt_pad<<<(int)((n+255)/256),256,0,stream>>>(Wih_f, WF, G4, EMB, EMB_P); }
  { long n = (long)G4 * EMB;  k_cvt_pad<<<(int)((n+255)/256),256,0,stream>>>(Wih_b, WB, G4, EMB, EMB_P); }
  { long n = (long)G4 * HID;  k_cvt_pad<<<(int)((n+255)/256),256,0,stream>>>(Whh_f, WHH, G4, HID, HID); }
  k_addb<<<16, 256, 0, stream>>>(bih_f, bhh_f, BSF, G4);
  k_addb<<<16, 256, 0, stream>>>(bih_b, bhh_b, BSB, G4);

  // 2) hoisted input-gate GEMMs (bias folded in)
  //    Xg_f[T*B][4096] = X @ Wih_f^T + bsum_f
  k_gemm<<<dim3((TSEQ*BATCH)/64, G4/64), 128, 0, stream>>>(
      X16, EMB_P, WF, EMB_P, XGF, G4, BSF, 0, EMB_P);
  //    backward needs only t = T-1 (first step of reversed scan)
  k_gemm<<<dim3(BATCH/64, G4/64), 128, 0, stream>>>(
      X16 + (size_t)(TSEQ - 1) * BATCH * EMB_P, EMB_P, WB, EMB_P, XGB, G4, BSB, 0, EMB_P);

  // 3) serial recurrent path: 45 fused WMMA+elementwise steps (ping-pong h)
  for (int t = 0; t < TSEQ; t++) {
    const __bf16* hp = (t & 1) ? H16B : H16A;
    __bf16*       hn = (t & 1) ? H16A : H16B;
    k_lstm_step<<<dim3(HID/16, 2), 128, 0, stream>>>(
        hp, WHH, XGF + (size_t)t * BATCH * G4, CBUF, hn, HF32);
  }
  k_lstm_b0<<<(BATCH*HID + 255) / 256, 256, 0, stream>>>(XGB, HB0);
  k_enc_norm<<<BATCH, 256, 0, stream>>>(HF32, HB0, enc_out);

  // 4) image branch
  k_imgnorm<<<PIX, 256, 0, stream>>>(img, VPAD);
  { long n = (long)CH * C0IN * 9; k_cvt_convw<<<(int)((n+255)/256),256,0,stream>>>(conv0_w, W0C, C0IN, C0IN_P); }
  k_conv3x3<<<(PIX/16)/4, 128, 0, stream>>>(VPAD, C0IN_P, W0C, conv0_b, OUTA);
  k_bnstats<<<CH, 256, 0, stream>>>(OUTA, MEAN, VAR);
  k_bnapply<<<(int)(((long)PIX*CH + 255)/256), 256, 0, stream>>>(
      OUTA, MEAN, VAR, bn0_g, bn0_b, nullptr, VCAT, nullptr);

  // ---- resblock 1 ----
  { long n = (long)CH * CCAT; k_cvt_pad<<<(int)((n+255)/256),256,0,stream>>>(r1c1_w, W1C, CH, CCAT, CCAT_P); }
  k_gemm<<<dim3(PIX/64, CH/64), 128, 0, stream>>>(          // v1 = relu(conv1)
      VCAT, CCAT_P, W1C, CCAT_P, OUTA, CH, r1c1_b, 1, CCAT_P);
  k_v1pad<<<(int)(((long)PIX*CH + 255)/256), 256, 0, stream>>>(OUTA, V1P);
  { long n = (long)CH * CH * 9; k_cvt_convw<<<(int)((n+255)/256),256,0,stream>>>(r1c2_w, W2C, CH, CH); }
  k_conv3x3<<<(PIX/16)/4, 128, 0, stream>>>(V1P, CH, W2C, r1c2_b, OUTB);
  k_bnstats<<<CH, 256, 0, stream>>>(OUTB, MEAN, VAR);
  k_bnapply<<<(int)(((long)PIX*CH + 255)/256), 256, 0, stream>>>(  // relu(bn)+v1 -> VCAT
      OUTB, MEAN, VAR, r1bn_g, r1bn_b, OUTA, VCAT, nullptr);

  // ---- resblock 2 ----
  { long n = (long)CH * CCAT; k_cvt_pad<<<(int)((n+255)/256),256,0,stream>>>(r2c1_w, W1C, CH, CCAT, CCAT_P); }
  k_gemm<<<dim3(PIX/64, CH/64), 128, 0, stream>>>(
      VCAT, CCAT_P, W1C, CCAT_P, OUTA, CH, r2c1_b, 1, CCAT_P);
  k_v1pad<<<(int)(((long)PIX*CH + 255)/256), 256, 0, stream>>>(OUTA, V1P);
  { long n = (long)CH * CH * 9; k_cvt_convw<<<(int)((n+255)/256),256,0,stream>>>(r2c2_w, W2C, CH, CH); }
  k_conv3x3<<<(PIX/16)/4, 128, 0, stream>>>(V1P, CH, W2C, r2c2_b, OUTB);
  k_bnstats<<<CH, 256, 0, stream>>>(OUTB, MEAN, VAR);
  k_bnapply<<<(int)(((long)PIX*CH + 255)/256), 256, 0, stream>>>(  // final -> d_out NCHW
      OUTB, MEAN, VAR, r2bn_g, r2bn_b, OUTA, nullptr, v_out);
}
